// MHAtt_80573586473609
// MI455X (gfx1250) — compile-verified, matrix operand order checked by reading, not run on hardware
//
#include <hip/hip_runtime.h>

typedef __bf16 bf16;
typedef __attribute__((ext_vector_type(16))) __bf16 v16bf;
typedef __attribute__((ext_vector_type(8)))  __bf16 v8bf;
typedef __attribute__((ext_vector_type(8)))  float  v8f;
typedef __attribute__((ext_vector_type(4)))  unsigned int v4u;
typedef __attribute__((ext_vector_type(8)))  int v8i;
typedef __attribute__((ext_vector_type(4)))  int v4i;

#define HIDDEN 512
#define NB 8
#define SEQ 1024
#define NH 8
#define HD 64

// workspace layout (bf16 element offsets). Total = 16M bf16 = 32 MB.
#define QH_OFF 0ull
#define KH_OFF 4194304ull
#define VT_OFF 8388608ull
#define AT_OFF 12582912ull

__device__ __forceinline__ v8f wmma_bf16(v16bf a, v16bf b, v8f c) {
  // D = A(16x32 bf16) x B(32x16 bf16) + C(16x16 f32)
  return __builtin_amdgcn_wmma_f32_16x16x32_bf16(false, a, false, b, (short)0, c,
                                                 false, false);
}

// A/B fragment from a row-major bf16 matrix (works for global or LDS pointers).
// Element j holds K = (j>>3)*16 + half*8 + (j&7).
__device__ __forceinline__ v16bf frag_bf16(const bf16* rowk0, int half) {
  union { v16bf v; v8bf h[2]; } u;
  u.h[0] = *(const v8bf*)(rowk0 + half * 8);
  u.h[1] = *(const v8bf*)(rowk0 + 16 + half * 8);
  return u.v;
}

// Same fragment sourced from f32 data (converted to bf16 in-register).
__device__ __forceinline__ v16bf frag_f32(const float* rowk0, int half) {
  v8f a = *(const v8f*)(rowk0 + half * 8);
  v8f b = *(const v8f*)(rowk0 + 16 + half * 8);
  v16bf r;
#pragma unroll
  for (int j = 0; j < 8; ++j) {
    r[j]     = (bf16)a[j];
    r[j + 8] = (bf16)b[j];
  }
  return r;
}

__device__ __forceinline__ float rmax16(float t) {
  t = fmaxf(t, __shfl_xor(t, 1, 32));
  t = fmaxf(t, __shfl_xor(t, 2, 32));
  t = fmaxf(t, __shfl_xor(t, 4, 32));
  t = fmaxf(t, __shfl_xor(t, 8, 32));
  return t;
}
__device__ __forceinline__ float rsum16(float t) {
  t += __shfl_xor(t, 1, 32);
  t += __shfl_xor(t, 2, 32);
  t += __shfl_xor(t, 4, 32);
  t += __shfl_xor(t, 8, 32);
  return t;
}

// Tensor Data Mover: 2-D tile (dim0 x dim1 elements of 2 bytes, row stride
// stride0 elements) from global -> LDS at byte offset ldsOff. D# layout per
// CDNA5 ISA (count=1, data_size=1 (2B), type=2).
__device__ __forceinline__ void tdm_load_2d(unsigned long long gaddr, unsigned ldsOff,
                                            unsigned dim0, unsigned dim1,
                                            unsigned stride0) {
  v4u g0;
  g0[0] = 1u;                                            // count = 1 (valid)
  g0[1] = ldsOff;                                        // lds_addr (bytes)
  g0[2] = (unsigned)(gaddr & 0xFFFFFFFFu);               // global_addr[31:0]
  g0[3] = (unsigned)((gaddr >> 32) & 0x01FFFFFFu) | 0x80000000u; // [56:32], type=2
  v8i g1;
  g1[0] = 0x00010000;                                    // data_size=1 -> 2 bytes
  g1[1] = (int)(dim0 << 16);                             // tensor_dim0[15:0]
  g1[2] = (int)((dim0 >> 16) | (dim1 << 16));            // tdim0[31:16] | tdim1[15:0]
  g1[3] = (int)((dim1 >> 16) | (dim0 << 16));            // tdim1[31:16] | tile_dim0
  g1[4] = (int)dim1;                                     // tile_dim1 (tile_dim2=0)
  g1[5] = (int)stride0;                                  // tensor_dim0_stride lo32
  g1[6] = 0;
  g1[7] = 0;
  v4i z = {0, 0, 0, 0};
#if __clang_major__ >= 23
  v8i z8 = {0, 0, 0, 0, 0, 0, 0, 0};
  __builtin_amdgcn_tensor_load_to_lds(g0, g1, z, z, z8, 0);
#else
  __builtin_amdgcn_tensor_load_to_lds(g0, g1, z, z, 0);
#endif
}

// ---------------------------------------------------------------------------
// Kernel 1: fused Q/K/V projections: Y = X @ W^T + b, stored as bf16.
// Q,K -> [bh][s][d]; V -> transposed [bh][d][s].
// One wave = 32x64 output tile (2 A-frags x 4 B-frags reused -> 8 WMMA/chunk,
// 48 B/lane of fragment traffic per WMMA instead of 128 B).
// ---------------------------------------------------------------------------
__global__ __launch_bounds__(256) void proj_qkv(
    const float* __restrict__ vin, const float* __restrict__ kin,
    const float* __restrict__ qin,
    const float* __restrict__ Wv, const float* __restrict__ bv,
    const float* __restrict__ Wk, const float* __restrict__ bk,
    const float* __restrict__ Wq, const float* __restrict__ bq,
    bf16* ws)
{
  const int lane = threadIdx.x & 31, m = lane & 15, half = lane >> 4;
  const int wave = threadIdx.x >> 5;
  const int tid   = blockIdx.x * 8 + wave;       // 6144 waves total
  const int matId = tid / 2048;                  // 0=Q 1=K 2=V
  const int rem   = tid % 2048;
  const int rt = rem >> 3, ct = rem & 7;         // 256 row-tiles(32) x 8 col-tiles(64)
  const float *X, *W, *bias;
  if (matId == 0)      { X = qin; W = Wq; bias = bq; }
  else if (matId == 1) { X = kin; W = Wk; bias = bk; }
  else                 { X = vin; W = Wv; bias = bv; }

  const int t0 = rt * 32, col0 = ct * 64;
  const float* arow0 = X + (size_t)(t0 + m) * HIDDEN;
  const float* arow1 = X + (size_t)(t0 + 16 + m) * HIDDEN;
  const float* brow0 = W + (size_t)(col0 + m) * HIDDEN;

  v8f acc[2][4] = {{{}, {}, {}, {}}, {{}, {}, {}, {}}};
#pragma unroll 2
  for (int c = 0; c < 16; ++c) {
    const v16bf a0 = frag_f32(arow0 + c * 32, half);
    const v16bf a1 = frag_f32(arow1 + c * 32, half);
#pragma unroll
    for (int ni = 0; ni < 4; ++ni) {
      const v16bf bfr = frag_f32(brow0 + (size_t)ni * 16 * HIDDEN + c * 32, half);
      acc[0][ni] = wmma_bf16(a0, bfr, acc[0][ni]);
      acc[1][ni] = wmma_bf16(a1, bfr, acc[1][ni]);
    }
  }

#pragma unroll
  for (int ni = 0; ni < 4; ++ni) {
    const float bb = bias[col0 + ni * 16 + m];
    const int f = col0 + ni * 16 + m, h = f >> 6, d = f & 63;
#pragma unroll
    for (int ri = 0; ri < 2; ++ri) {
      if (matId < 2) {
        bf16* out = ws + (matId == 0 ? QH_OFF : KH_OFF);
#pragma unroll
        for (int r = 0; r < 8; ++r) {
          const int t = t0 + ri * 16 + half * 8 + r;
          const int b = t >> 10, s = t & 1023;
          out[(((size_t)(b * NH + h)) * SEQ + s) * HD + d] = (bf16)(acc[ri][ni][r] + bb);
        }
      } else {
        bf16* out = ws + VT_OFF;
        const int b = t0 >> 10, s0 = (t0 & 1023) + ri * 16 + half * 8;
        v8bf pk;
#pragma unroll
        for (int r = 0; r < 8; ++r) pk[r] = (bf16)(acc[ri][ni][r] + bb);
        *(v8bf*)(out + (((size_t)(b * NH + h)) * HD + d) * SEQ + s0) = pk;
      }
    }
  }
}

// ---------------------------------------------------------------------------
// Kernel 2: attention. Block = one (b,h); its 8 waves each own a 16-query-row
// tile. Wave 0 TDM-DMAs the whole K head matrix (128 KB bf16) into LDS once;
// both score passes then read K via ds_load. Pass 1: online softmax stats.
// Pass 2: recompute scores, single normalized f32 att_map write, C->A LDS
// transpose, att*V WMMAs against V^T (global, L2-resident, read once).
// ---------------------------------------------------------------------------
__global__ __launch_bounds__(256) void attn(
    const bf16* ws, const unsigned char* __restrict__ mask,
    float* __restrict__ amap_all, bf16* atted)
{
  struct Smem {
    bf16 k[SEQ * HD];        // 128 KB: full K head matrix
    bf16 pbuf[8][16][32];    // 8 KB: per-wave prob chunk (C->A transpose)
  };
  __shared__ Smem smem;

  const int lane = threadIdx.x & 31, m = lane & 15, half = lane >> 4;
  const int wave = threadIdx.x >> 5;
  const int bh = blockIdx.x >> 3;                // 64 (b,h) x 8 q-groups
  const int qt = (blockIdx.x & 7) * 8 + wave;
  const int b = bh >> 3, h = bh & 7;

  const bf16* qh = ws + QH_OFF + (size_t)bh * SEQ * HD;
  const bf16* vt = ws + VT_OFF + (size_t)bh * HD * SEQ;
  const unsigned char* mrow = mask + (size_t)b * SEQ;

  if (wave == 0) {
    const unsigned long long ka =
        (unsigned long long)(size_t)(ws + KH_OFF + (size_t)bh * SEQ * HD);
    tdm_load_2d(ka, 0u /*lds byte offset of smem.k*/, HD, SEQ, HD);
    __builtin_amdgcn_s_wait_tensorcnt(0);
  }
  __syncthreads();

  // Q fragments (16 rows x 64 d), kept in registers for both passes.
  const bf16* qrow = qh + (size_t)(qt * 16 + m) * HD;
  const v16bf qa0 = frag_bf16(qrow, half);
  const v16bf qa1 = frag_bf16(qrow + 32, half);

  float M[8], L[8];
#pragma unroll
  for (int r = 0; r < 8; ++r) { M[r] = -3.0e38f; L[r] = 0.0f; }

  // ---- pass 1: online row max / sum-of-exp over all 64 key tiles ----
  for (int kt = 0; kt < 64; ++kt) {
    const bf16* krow = smem.k + (size_t)(kt * 16 + m) * HD;   // LDS
    v8f acc = {};
    acc = wmma_bf16(qa0, frag_bf16(krow, half), acc);
    acc = wmma_bf16(qa1, frag_bf16(krow + 32, half), acc);
    const bool mk = mrow[kt * 16 + m] != 0;                   // column (=key)
#pragma unroll
    for (int r = 0; r < 8; ++r) {
      float s = acc[r] * 0.125f;                              // 1/sqrt(64)
      if (mk) s = -10000.0f;
      const float tmax = rmax16(s);
      const float newM = fmaxf(M[r], tmax);
      const float se = rsum16(__expf(s - newM));
      L[r] = L[r] * __expf(M[r] - newM) + se;
      M[r] = newM;
    }
  }
  float invL[8];
#pragma unroll
  for (int r = 0; r < 8; ++r) invL[r] = 1.0f / L[r];

  // ---- pass 2: recompute scores, emit probs, accumulate att*V ----
  v8f o[4] = {{}, {}, {}, {}};
  float* amap = amap_all + ((size_t)bh * SEQ + qt * 16) * SEQ;
  for (int kc = 0; kc < 32; ++kc) {                // 32 key-chunks of 32
#pragma unroll
    for (int sub = 0; sub < 2; ++sub) {
      const int kt = kc * 2 + sub;
      const bf16* krow = smem.k + (size_t)(kt * 16 + m) * HD; // LDS
      v8f acc = {};
      acc = wmma_bf16(qa0, frag_bf16(krow, half), acc);
      acc = wmma_bf16(qa1, frag_bf16(krow + 32, half), acc);
      const bool mk = mrow[kt * 16 + m] != 0;
#pragma unroll
      for (int r = 0; r < 8; ++r) {
        float s = acc[r] * 0.125f;
        if (mk) s = -10000.0f;
        const float p = __expf(s - M[r]) * invL[r];
        amap[(size_t)(half * 8 + r) * SEQ + kt * 16 + m] = p;  // f32 att_map
        smem.pbuf[wave][half * 8 + r][sub * 16 + m] = (bf16)p; // C->A transpose
      }
    }
    asm volatile("s_wait_dscnt 0x0" ::: "memory"); // in-wave LDS store->load
    union { v16bf v; v8bf hh[2]; } pa;
    pa.hh[0] = *(const v8bf*)&smem.pbuf[wave][m][half * 8];
    pa.hh[1] = *(const v8bf*)&smem.pbuf[wave][m][16 + half * 8];
#pragma unroll
    for (int nt = 0; nt < 4; ++nt) {
      // B col n = d-column -> row of V^T, contiguous over keys
      const v16bf vb = frag_bf16(vt + (size_t)(nt * 16 + m) * SEQ + kc * 32, half);
      o[nt] = wmma_bf16(pa.v, vb, o[nt]);
    }
  }

  bf16* ao = atted + ((size_t)(b * SEQ + qt * 16)) * HIDDEN + h * HD;
#pragma unroll
  for (int nt = 0; nt < 4; ++nt)
#pragma unroll
    for (int r = 0; r < 8; ++r)
      ao[(size_t)(half * 8 + r) * HIDDEN + nt * 16 + m] = (bf16)o[nt][r];
}

// ---------------------------------------------------------------------------
// Kernel 3: output projection atted(bf16) @ Wm^T + bm -> f32 d_out.
// One wave = 32x64 output tile.
// ---------------------------------------------------------------------------
__global__ __launch_bounds__(256) void proj_out(
    const bf16* atted, const float* __restrict__ Wm,
    const float* __restrict__ bm, float* __restrict__ out)
{
  const int lane = threadIdx.x & 31, m = lane & 15, half = lane >> 4;
  const int wave = threadIdx.x >> 5;
  const int tid = blockIdx.x * 8 + wave;         // 2048 waves
  const int rt = tid >> 3, ct = tid & 7;
  const int t0 = rt * 32, col0 = ct * 64;

  const bf16*  arow0 = atted + (size_t)(t0 + m) * HIDDEN;
  const bf16*  arow1 = atted + (size_t)(t0 + 16 + m) * HIDDEN;
  const float* brow0 = Wm + (size_t)(col0 + m) * HIDDEN;

  v8f acc[2][4] = {{{}, {}, {}, {}}, {{}, {}, {}, {}}};
#pragma unroll 2
  for (int c = 0; c < 16; ++c) {
    const v16bf a0 = frag_bf16(arow0 + c * 32, half);
    const v16bf a1 = frag_bf16(arow1 + c * 32, half);
#pragma unroll
    for (int ni = 0; ni < 4; ++ni) {
      const v16bf bfr = frag_f32(brow0 + (size_t)ni * 16 * HIDDEN + c * 32, half);
      acc[0][ni] = wmma_bf16(a0, bfr, acc[0][ni]);
      acc[1][ni] = wmma_bf16(a1, bfr, acc[1][ni]);
    }
  }
#pragma unroll
  for (int ni = 0; ni < 4; ++ni) {
    const float bb = bm[col0 + ni * 16 + m];
#pragma unroll
    for (int ri = 0; ri < 2; ++ri)
#pragma unroll
      for (int r = 0; r < 8; ++r)
        out[(size_t)(t0 + ri * 16 + half * 8 + r) * HIDDEN + col0 + ni * 16 + m] =
            acc[ri][ni][r] + bb;
  }
}

extern "C" void kernel_launch(void* const* d_in, const int* in_sizes, int n_in,
                              void* d_out, int out_size, void* d_ws, size_t ws_size,
                              hipStream_t stream) {
  (void)in_sizes; (void)n_in; (void)out_size; (void)ws_size;
  const float* v  = (const float*)d_in[0];
  const float* k  = (const float*)d_in[1];
  const float* q  = (const float*)d_in[2];
  const unsigned char* mask = (const unsigned char*)d_in[3];
  const float* Wv = (const float*)d_in[4];
  const float* bv = (const float*)d_in[5];
  const float* Wk = (const float*)d_in[6];
  const float* bk = (const float*)d_in[7];
  const float* Wq = (const float*)d_in[8];
  const float* bq = (const float*)d_in[9];
  const float* Wm = (const float*)d_in[10];
  const float* bm = (const float*)d_in[11];

  float* out  = (float*)d_out;
  float* amap = out + (size_t)NB * SEQ * HIDDEN;   // att_map region of d_out
  bf16*  ws   = (bf16*)d_ws;                        // needs 32 MB

  proj_qkv<<<768, 256, 0, stream>>>(v, k, q, Wv, bv, Wk, bk, Wq, bq, ws);
  attn<<<512, 256, 0, stream>>>(ws, mask, amap, ws + AT_OFF);
  proj_out<<<256, 256, 0, stream>>>(ws + AT_OFF, Wm, bm, out);
}
